// SGMNet_90623809946416
// MI455X (gfx1250) — matrix operand order repaired and checked by reading, not compiled, not run on hardware
//
#include <hip/hip_runtime.h>

// ---------------------------------------------------------------------------
// Constants
// ---------------------------------------------------------------------------
#define BATCH 4
#define NPTS 2048
#define CH 256
#define NH 4
#define HDIM 64
#define NLAYERS 9
#define SEED 128
#define MIDL 6
#define NBD 2049          // NPTS + dustbin
#define EPS_ 1e-8f
#define NMS_R 0.01f

// GEMM tiling: 64x64 block tile, K-slab of 64 = two WMMA K-steps per stage.
#define BM 64
#define BN 64
#define BK 64
#define LDA_S (BK + 8)    // 72 halves -> 144B rows (multiple of 16B)
#define LDB_T (BK + 8)

// Flip to 0 if the assembler rejects the async-to-LDS mnemonic.
#ifndef USE_ASYNC
#define USE_ASYNC 1
#endif

typedef __attribute__((ext_vector_type(16))) _Float16 v16h;
typedef __attribute__((ext_vector_type(8)))  float    v8f;
typedef __attribute__((ext_vector_type(4)))  float    v4f;
typedef __attribute__((ext_vector_type(4)))  _Float16 h4;

union H16 { v16h v; _Float16 h[16]; };
union F8  { v8f  v; float    f[8];  };

// ---------------------------------------------------------------------------
// WMMA fragment loaders from LDS (CDNA5 16x16x32 f16 layouts, CDNA5 ISA)
// A [BM][LDA_S]: lane m holds K-halves {kb..kb+7, 16+kb..16+kb+7}, kb=(lane>>4)*8
// B stored transposed [BN][LDB_T]: lane n holds K-halves {kb..kb+7, 16+kb..16+kb+7}, kb=(lane>>4)*8
// Both loaders read two 16B-aligned runs -> ds_load_b128 pairs.
// ---------------------------------------------------------------------------
__device__ inline v16h frag_a(const _Float16* As, int ld) {
  int lane = threadIdx.x & 31;
  int m = lane & 15;
  int kb = (lane >> 4) * 8;
  const _Float16* r = As + m * ld;
  H16 t;
#pragma unroll
  for (int e = 0; e < 8; ++e) t.h[e] = r[kb + e];
#pragma unroll
  for (int e = 0; e < 8; ++e) t.h[8 + e] = r[16 + kb + e];
  return t.v;
}

__device__ inline v16h frag_bT(const _Float16* BsT, int ld) {
  int lane = threadIdx.x & 31;
  int n = lane & 15;
  int kb = (lane >> 4) * 8;
  const _Float16* r = BsT + n * ld;
  H16 t;
#pragma unroll
  for (int e = 0; e < 8; ++e) t.h[e] = r[kb + e];
#pragma unroll
  for (int e = 0; e < 8; ++e) t.h[8 + e] = r[16 + kb + e];
  return t.v;
}

// Async 16B global -> LDS copy (CDNA5 VFLAT async op, tracked by ASYNCcnt).
__device__ inline void async_g2l_b128(unsigned lds_addr, const float* gp) {
#if USE_ASYNC
  asm volatile("global_load_async_to_lds_b128 %0, %1, off"
               :: "v"(lds_addr), "v"((unsigned long long)gp) : "memory");
#else
  (void)lds_addr; (void)gp;
#endif
}
__device__ inline void wait_async0() {
#if USE_ASYNC
  asm volatile("s_wait_asynccnt 0x0" ::: "memory");
#endif
}
__device__ inline unsigned lds_addr_of(const void* p) {
  return (unsigned)(unsigned long long)p;   // low 32 bits of flat LDS address
}

// ---------------------------------------------------------------------------
// General batched WMMA GEMM:  C = alpha * op(A) * op(B)  [+bias][+res][+C][relu][dist]
// grid.z = batch * ZH; offsets: zb*s?b + zh*s?h. 128 threads = 4 waves, each
// wave computes a 32x32 quadrant = 2x2 WMMA 16x16 tiles, 2 K-steps per stage.
// Full interior tiles: async-DMA f32 tiles to LDS, convert to f16 vectors.
// Edge tiles: scalar bounds-checked fallback.
// ---------------------------------------------------------------------------
__global__ __launch_bounds__(128)
void k_gemm(const float* __restrict__ A, const float* __restrict__ Bm,
            float* __restrict__ Cc,
            const float* __restrict__ bias, const float* __restrict__ res,
            int M, int Nn, int K, int lda, int ldb, int ldc, int ldr,
            long sAb, long sAh, long sBb, long sBh, long sCb, long sCh, long sRb,
            int ZH, int transA, int transB, float alpha,
            int relu, int accum, int dist)
{
  __shared__ float    AsF[BM * BK];         // f32 staging (layout depends on transA)
  __shared__ float    BsF[BK * BN];         // f32 staging (layout depends on transB)
  __shared__ _Float16 As[BM * LDA_S];       // f16 [BM][BK]
  __shared__ _Float16 Bs[BN * LDB_T];       // f16 transposed [BN][BK]

  int z = blockIdx.z, zb = z / ZH, zh = z % ZH;
  A  += zb * sAb + zh * sAh;
  Bm += zb * sBb + zh * sBh;
  Cc += zb * sCb + zh * sCh;
  if (res) res += zb * sRb;

  int m0 = blockIdx.y * BM, n0 = blockIdx.x * BN;
  int tid = threadIdx.x, wave = tid >> 5;
  int wm = (wave & 1) * 32, wn = (wave >> 1) * 32;

  unsigned lbA = lds_addr_of(AsF);
  unsigned lbB = lds_addr_of(BsF);

  F8 zz;
#pragma unroll
  for (int e = 0; e < 8; ++e) zz.f[e] = 0.f;
  v8f acc00 = zz.v, acc01 = zz.v, acc10 = zz.v, acc11 = zz.v;

  for (int k0 = 0; k0 < K; k0 += BK) {
    bool full = (m0 + BM <= M) && (n0 + BN <= Nn) && (k0 + BK <= K);
    if (full) {
      // ---- phase 1: async DMA f32 tiles into LDS staging ----
      // All four layouts are 64 rows x 16 float4-groups.
      if (!transA) {                       // AsF layout [BM][BK], contiguous along k
        for (int g = tid; g < (BM * BK) / 4; g += 128) {
          int r = g >> 4, c4 = (g & 15) * 4;
          async_g2l_b128(lbA + (unsigned)((r * BK + c4) * 4),
                         A + (long)(m0 + r) * lda + (k0 + c4));
        }
      } else {                             // AsF layout [BK][BM], contiguous along m
        for (int g = tid; g < (BK * BM) / 4; g += 128) {
          int r = g >> 4, c4 = (g & 15) * 4;
          async_g2l_b128(lbA + (unsigned)((r * BM + c4) * 4),
                         A + (long)(k0 + r) * lda + (m0 + c4));
        }
      }
      if (!transB) {                       // BsF layout [BK][BN], contiguous along n
        for (int g = tid; g < (BK * BN) / 4; g += 128) {
          int r = g >> 4, c4 = (g & 15) * 4;
          async_g2l_b128(lbB + (unsigned)((r * BN + c4) * 4),
                         Bm + (long)(k0 + r) * ldb + (n0 + c4));
        }
      } else {                             // BsF layout [BN][BK], contiguous along k
        for (int g = tid; g < (BN * BK) / 4; g += 128) {
          int r = g >> 4, c4 = (g & 15) * 4;
          async_g2l_b128(lbB + (unsigned)((r * BK + c4) * 4),
                         Bm + (long)(n0 + r) * ldb + (k0 + c4));
        }
      }
#if !USE_ASYNC
      if (!transA) {
        for (int g = tid; g < (BM * BK) / 4; g += 128) {
          int r = g >> 4, c4 = (g & 15) * 4;
          *(v4f*)&AsF[r * BK + c4] = *(const v4f*)(A + (long)(m0 + r) * lda + (k0 + c4));
        }
      } else {
        for (int g = tid; g < (BK * BM) / 4; g += 128) {
          int r = g >> 4, c4 = (g & 15) * 4;
          *(v4f*)&AsF[r * BM + c4] = *(const v4f*)(A + (long)(k0 + r) * lda + (m0 + c4));
        }
      }
      if (!transB) {
        for (int g = tid; g < (BK * BN) / 4; g += 128) {
          int r = g >> 4, c4 = (g & 15) * 4;
          *(v4f*)&BsF[r * BN + c4] = *(const v4f*)(Bm + (long)(k0 + r) * ldb + (n0 + c4));
        }
      } else {
        for (int g = tid; g < (BN * BK) / 4; g += 128) {
          int r = g >> 4, c4 = (g & 15) * 4;
          *(v4f*)&BsF[r * BK + c4] = *(const v4f*)(Bm + (long)(n0 + r) * ldb + (k0 + c4));
        }
      }
#endif
      wait_async0();
      __syncthreads();
      // ---- phase 2: LDS f32 -> LDS f16 (A row-major, B transposed) ----
      if (!transA) {
        for (int g = tid; g < (BM * BK) / 4; g += 128) {
          int r = g >> 4, c4 = (g & 15) * 4;
          v4f f = *(const v4f*)&AsF[r * BK + c4];
          *(h4*)&As[r * LDA_S + c4] = __builtin_convertvector(f, h4);
        }
      } else {
        for (int i = tid; i < BM * BK; i += 128) {
          int m = i >> 6, k = i & 63;
          As[m * LDA_S + k] = (_Float16)AsF[k * BM + m];
        }
      }
      if (transB) {
        for (int g = tid; g < (BN * BK) / 4; g += 128) {
          int r = g >> 4, c4 = (g & 15) * 4;
          v4f f = *(const v4f*)&BsF[r * BK + c4];
          *(h4*)&Bs[r * LDB_T + c4] = __builtin_convertvector(f, h4);
        }
      } else {
        for (int i = tid; i < BK * BN; i += 128) {
          int k = i >> 6, n = i & 63;
          Bs[n * LDB_T + k] = (_Float16)BsF[k * BN + n];
        }
      }
    } else {
      // ---- edge tiles: scalar bounds-checked fallback ----
      for (int i = tid; i < BM * BK; i += 128) {
        int r = i >> 6, c = i & 63;
        int gm = m0 + r, gk = k0 + c;
        float v = 0.f;
        if (gm < M && gk < K) v = transA ? A[(long)gk * lda + gm] : A[(long)gm * lda + gk];
        As[r * LDA_S + c] = (_Float16)v;
      }
      for (int i = tid; i < BK * BN; i += 128) {
        int r = i >> 6, c = i & 63;   // r = k, c = n
        int gk = k0 + r, gn = n0 + c;
        float v = 0.f;
        if (gk < K && gn < Nn) v = transB ? Bm[(long)gn * ldb + gk] : Bm[(long)gk * ldb + gn];
        Bs[c * LDB_T + r] = (_Float16)v;
      }
    }
    __syncthreads();

#pragma unroll
    for (int ks = 0; ks < BK; ks += 32) {
      v16h a0 = frag_a(&As[(wm + 0)  * LDA_S + ks], LDA_S);
      v16h a1 = frag_a(&As[(wm + 16) * LDA_S + ks], LDA_S);
      v16h b0 = frag_bT(&Bs[(wn + 0)  * LDB_T + ks], LDB_T);
      v16h b1 = frag_bT(&Bs[(wn + 16) * LDB_T + ks], LDB_T);
      acc00 = __builtin_amdgcn_wmma_f32_16x16x32_f16(false, a0, false, b0, (short)0, acc00, false, false);
      acc01 = __builtin_amdgcn_wmma_f32_16x16x32_f16(false, a0, false, b1, (short)0, acc01, false, false);
      acc10 = __builtin_amdgcn_wmma_f32_16x16x32_f16(false, a1, false, b0, (short)0, acc10, false, false);
      acc11 = __builtin_amdgcn_wmma_f32_16x16x32_f16(false, a1, false, b1, (short)0, acc11, false, false);
    }
    __syncthreads();
  }

  int lane = tid & 31;
  int cn = lane & 15, chh = lane >> 4;
  v8f accs[2][2] = { { acc00, acc01 }, { acc10, acc11 } };
#pragma unroll
  for (int i = 0; i < 2; ++i)
#pragma unroll
    for (int j = 0; j < 2; ++j) {
      F8 u; u.v = accs[i][j];
#pragma unroll
      for (int r = 0; r < 8; ++r) {
        int gm = m0 + wm + 16 * i + 8 * chh + r;
        int gn = n0 + wn + 16 * j + cn;
        if (gm < M && gn < Nn) {
          float v = alpha * u.f[r];
          if (dist) v = sqrtf(fmaxf(2.f - 2.f * v, 1e-12f));
          if (bias) v += bias[gm];
          if (res)  v += res[(long)gm * ldr + gn];
          long o = (long)gm * ldc + gn;
          if (accum) v += Cc[o];
          if (relu) v = fmaxf(v, 0.f);
          Cc[o] = v;
        }
      }
    }
}

// ---------------------------------------------------------------------------
// Reduction helper (blockDim == 256)
// ---------------------------------------------------------------------------
__device__ inline float blk_sum(float v, float* sh) {
  int t = threadIdx.x;
  sh[t] = v; __syncthreads();
  for (int s = 128; s > 0; s >>= 1) { if (t < s) sh[t] += sh[t + s]; __syncthreads(); }
  float r = sh[0]; __syncthreads();
  return r;
}

// ---------------------------------------------------------------------------
// Elementwise / reduction kernels
// ---------------------------------------------------------------------------
__global__ void k_softmax_rows(float* X, int L, int ld) {
  __shared__ float sh[256];
  float* p = X + (long)blockIdx.x * ld;
  int t = threadIdx.x;
  float mx = -1e30f;
  for (int i = t; i < L; i += 256) mx = fmaxf(mx, p[i]);
  sh[t] = mx; __syncthreads();
  for (int s = 128; s > 0; s >>= 1) { if (t < s) sh[t] = fmaxf(sh[t], sh[t + s]); __syncthreads(); }
  mx = sh[0]; __syncthreads();
  float sum = 0.f;
  for (int i = t; i < L; i += 256) { float e = __expf(p[i] - mx); p[i] = e; sum += e; }
  sum = blk_sum(sum, sh);
  float inv = 1.f / sum;
  for (int i = t; i < L; i += 256) p[i] *= inv;
}

__global__ void k_bnrelu(float* X, const float* g, const float* bb,
                         const float* rm, const float* rv, int ch, int n, long tot) {
  long i = blockIdx.x * 256L + threadIdx.x;
  if (i >= tot) return;
  int c = (int)((i / n) % ch);
  float sc = g[c] * rsqrtf(rv[c] + 1e-5f);
  float sh = bb[c] - rm[c] * sc;
  X[i] = fmaxf(X[i] * sc + sh, 0.f);
}

__global__ void k_inorm_bnrelu(const float* X, float* Y, const float* g, const float* bb,
                               const float* rm, const float* rv, int ch, int n) {
  __shared__ float sh[256];
  long row = blockIdx.x;
  int c = (int)(row % ch);
  const float* x = X + row * (long)n;
  float* y = Y + row * (long)n;
  int t = threadIdx.x;
  float s = 0.f, ss = 0.f;
  for (int i = t; i < n; i += 256) { float v = x[i]; s += v; ss += v * v; }
  s = blk_sum(s, sh);
  ss = blk_sum(ss, sh);
  float mean = s / n;
  float var = ss / n - mean * mean;
  float inv = rsqrtf(var + 1e-3f);
  float sc = g[c] * rsqrtf(rv[c] + 1e-5f);
  float shf = bb[c] - rm[c] * sc;
  for (int i = t; i < n; i += 256) y[i] = fmaxf((x[i] - mean) * inv * sc + shf, 0.f);
}

__global__ void k_copy_ch(float* dst, const float* src, long chn, long sD, long sS, long tot) {
  long i = blockIdx.x * 256L + threadIdx.x;
  if (i >= tot) return;
  long b = i / chn, r = i % chn;
  dst[b * sD + r] = src[b * sS + r];
}

__global__ void k_gather_cols(float* dst, const float* src, const int* sel, int nsrc, long sSrc) {
  long i = blockIdx.x * 256L + threadIdx.x;
  long tot = (long)BATCH * CH * SEED;
  if (i >= tot) return;
  int b = (int)(i / ((long)CH * SEED));
  long r = i % ((long)CH * SEED);
  int c = (int)(r / SEED), j = (int)(r % SEED);
  dst[i] = src[(long)b * sSrc + (long)c * nsrc + sel[b * SEED + j]];
}

__global__ void k_mulcols(float* V, const float* w, int n2, long tot) {
  long i = blockIdx.x * 256L + threadIdx.x;
  if (i >= tot) return;
  int b = (int)(i / ((long)CH * n2));
  int j = (int)(i % n2);
  V[i] *= w[b * n2 + j];
}

__global__ void k_sigmoid(const float* x, float* y, long tot) {
  long i = blockIdx.x * 256L + threadIdx.x;
  if (i >= tot) return;
  y[i] = 1.f / (1.f + __expf(-x[i]));
}

__global__ void k_set(float* p, float v, long tot) {
  long i = blockIdx.x * 256L + threadIdx.x;
  if (i < tot) p[i] = v;
}

// kpts -> normalized coords, [B,2,N] (for conv) and [B,N,2] (for seeding)
__global__ void k_xt(const float* kp, float* xt, float* xs) {
  long i = blockIdx.x * 256L + threadIdx.x;
  long tot = (long)BATCH * NPTS;
  if (i >= tot) return;
  int b = (int)(i / NPTS), n = (int)(i % NPTS);
  float x = (kp[i * 2 + 0] - 640.f) / 896.f;   // shift=(1280,960)/2, scale=1280*0.7
  float y = (kp[i * 2 + 1] - 480.f) / 896.f;
  xt[(long)b * 2 * NPTS + n] = x;
  xt[(long)b * 2 * NPTS + NPTS + n] = y;
  xs[i * 2 + 0] = x;
  xs[i * 2 + 1] = y;
}

// normalize descriptor rows; emit [B,N,C] and transposed [B,C,N]
__global__ void k_normdesc(const float* d, float* dn, float* dt) {
  __shared__ float sh[256];
  int b = blockIdx.y, n = blockIdx.x, c = threadIdx.x;
  float v = d[((long)b * NPTS + n) * CH + c];
  float ss = blk_sum(v * v, sh);
  float nrm = fmaxf(sqrtf(ss), 1e-12f);
  float o = v / nrm;
  dn[((long)b * NPTS + n) * CH + c] = o;
  dt[((long)b * CH + c) * NPTS + n] = o;
}

// two smallest of a distance row + argmin
__global__ void k_top2(const float* dis, float* v0o, float* v1o, int* nn1) {
  int b = blockIdx.y, n = blockIdx.x, t = threadIdx.x;
  const float* row = dis + ((long)b * NPTS + n) * NPTS;
  float b0 = 1e30f, b1 = 1e30f; int i0 = 0;
  for (int m = t; m < NPTS; m += 256) {
    float v = row[m];
    if (v < b0) { b1 = b0; b0 = v; i0 = m; }
    else if (v < b1) b1 = v;
  }
  __shared__ float s0[256], s1[256]; __shared__ int si[256];
  s0[t] = b0; s1[t] = b1; si[t] = i0; __syncthreads();
  for (int s = 128; s > 0; s >>= 1) {
    if (t < s) {
      float a0 = s0[t], a1 = s1[t]; int ai = si[t];
      float c0 = s0[t + s], c1 = s1[t + s]; int ci = si[t + s];
      if (c0 < a0 || (c0 == a0 && ci < ai)) { s0[t] = c0; si[t] = ci; s1[t] = fminf(a0, c1); }
      else { s1[t] = fminf(a1, c0); }
    }
    __syncthreads();
  }
  if (t == 0) { v0o[b * NPTS + n] = s0[0]; v1o[b * NPTS + n] = s1[0]; nn1[b * NPTS + n] = si[0]; }
}

__global__ void k_colargmin(const float* dis, int* nn2) {
  int b = blockIdx.y, m = blockIdx.x, t = threadIdx.x;
  const float* col = dis + (long)b * NPTS * NPTS + m;
  float best = 1e30f; int bi = 0;
  for (int n = t; n < NPTS; n += 256) { float v = col[(long)n * NPTS]; if (v < best) { best = v; bi = n; } }
  __shared__ float sv[256]; __shared__ int si[256];
  sv[t] = best; si[t] = bi; __syncthreads();
  for (int s = 128; s > 0; s >>= 1) {
    if (t < s) {
      if (sv[t + s] < sv[t] || (sv[t + s] == sv[t] && si[t + s] < si[t])) { sv[t] = sv[t + s]; si[t] = si[t + s]; }
    }
    __syncthreads();
  }
  if (t == 0) nn2[b * NPTS + m] = si[0];
}

__global__ void k_ratio(const float* v0, const float* v1, float* r, long tot) {
  long i = blockIdx.x * 256L + threadIdx.x;
  if (i < tot) r[i] = v1[i] / (v0[i] + EPS_);
}

__global__ void k_gatherxy(const float* xs2, const int* nn1, float* x2g) {
  long i = blockIdx.x * 256L + threadIdx.x;
  long tot = (long)BATCH * NPTS;
  if (i >= tot) return;
  int b = (int)(i / NPTS);
  int j = nn1[i];
  x2g[i * 2 + 0] = xs2[((long)b * NPTS + j) * 2 + 0];
  x2g[i * 2 + 1] = xs2[((long)b * NPTS + j) * 2 + 1];
}

__global__ void k_pdist_sum(const float* xs, float* acc, int off) {
  __shared__ float sh[256];
  int b = blockIdx.y, n = blockIdx.x, t = threadIdx.x;
  const float* X = xs + (long)b * NPTS * 2;
  float ax = X[2 * n], ay = X[2 * n + 1];
  float s = 0.f;
  for (int m = t; m < NPTS; m += 256) {
    float dx = ax - X[2 * m], dy = ay - X[2 * m + 1];
    s += sqrtf(dx * dx + dy * dy);
  }
  s = blk_sum(s, sh);
  if (t == 0) atomicAdd(&acc[off + b], s);
}

__global__ void k_mutual(const int* nn1, const int* nn2, const float* score, float* sf) {
  long i = blockIdx.x * 256L + threadIdx.x;
  long tot = (long)BATCH * NPTS;
  if (i >= tot) return;
  int b = (int)(i / NPTS), n = (int)(i % NPTS);
  int j = nn1[i];
  sf[i] = (nn2[(long)b * NPTS + j] == n) ? score[i] : -1.f;
}

__global__ void k_ismax(const float* SF, const float* xs1, const float* x2g,
                        const float* mean, float* SEL, float confbar) {
  __shared__ int red[256];
  int b = blockIdx.y, n = blockIdx.x, t = threadIdx.x;
  float inv = 1.f / ((float)NPTS * (float)NPTS);
  float r1 = NMS_R * mean[b] * inv;
  float r2 = NMS_R * mean[BATCH + b] * inv;
  const float* X1 = xs1 + (long)b * NPTS * 2;
  const float* X2 = x2g + (long)b * NPTS * 2;
  float ax = X1[2 * n], ay = X1[2 * n + 1], cx = X2[2 * n], cy = X2[2 * n + 1];
  float sn = SF[(long)b * NPTS + n];
  int ok = 1;
  for (int m = t; m < NPTS; m += 256) {
    float dx = ax - X1[2 * m], dy = ay - X1[2 * m + 1];
    float d1 = sqrtf(dx * dx + dy * dy);
    float ex = cx - X2[2 * m], ey = cy - X2[2 * m + 1];
    float d2 = sqrtf(ex * ex + ey * ey);
    int nms = (d1 >= r1) && (d2 >= r2);
    if (!((sn >= SF[(long)b * NPTS + m]) || nms)) ok = 0;
  }
  red[t] = ok; __syncthreads();
  for (int s = 128; s > 0; s >>= 1) { if (t < s) red[t] &= red[t + s]; __syncthreads(); }
  if (t == 0) {
    float v = red[0] ? sn : -1.f;
    if (v < confbar) v = -1.f;
    SEL[(long)b * NPTS + n] = v;
  }
}

__global__ void k_topk(float* sel, const int* nn1, int* s1, int* s2) {
  __shared__ float bv[256]; __shared__ int bi[256];
  int b = blockIdx.x, t = threadIdx.x;
  float* sc = sel + (long)b * NPTS;
  for (int it = 0; it < SEED; ++it) {
    float best = -1e30f; int bidx = 0;
    for (int n = t; n < NPTS; n += 256) { float v = sc[n]; if (v > best) { best = v; bidx = n; } }
    bv[t] = best; bi[t] = bidx; __syncthreads();
    for (int s = 128; s > 0; s >>= 1) {
      if (t < s) {
        if (bv[t + s] > bv[t] || (bv[t + s] == bv[t] && bi[t + s] < bi[t])) { bv[t] = bv[t + s]; bi[t] = bi[t + s]; }
      }
      __syncthreads();
    }
    if (t == 0) {
      int idx = bi[0];
      s1[b * SEED + it] = idx;
      s2[b * SEED + it] = nn1[(long)b * NPTS + idx];
      sc[idx] = -1e30f;
    }
    __syncthreads();
  }
}

__global__ void k_rowmax_arg(const float* P, float* msc, int* nn1) {
  __shared__ float sv[256]; __shared__ int si[256];
  int b = blockIdx.y, n = blockIdx.x, t = threadIdx.x;
  const float* p = P + ((long)b * NBD + n) * NBD;
  float best = -1e30f; int bi = 0;
  for (int m = t; m < NPTS; m += 256) { float v = p[m]; if (v > best) { best = v; bi = m; } }
  sv[t] = best; si[t] = bi; __syncthreads();
  for (int s = 128; s > 0; s >>= 1) {
    if (t < s) {
      if (sv[t + s] > sv[t] || (sv[t + s] == sv[t] && si[t + s] < si[t])) { sv[t] = sv[t + s]; si[t] = si[t + s]; }
    }
    __syncthreads();
  }
  if (t == 0) { msc[(long)b * NPTS + n] = sv[0]; nn1[(long)b * NPTS + n] = si[0]; }
}

__global__ void k_colmax_arg(const float* P, int* nn2) {
  __shared__ float sv[256]; __shared__ int si[256];
  int b = blockIdx.y, m = blockIdx.x, t = threadIdx.x;
  const float* p = P + (long)b * NBD * NBD + m;
  float best = -1e30f; int bi = 0;
  for (int n = t; n < NPTS; n += 256) { float v = p[(long)n * NBD]; if (v > best) { best = v; bi = n; } }
  sv[t] = best; si[t] = bi; __syncthreads();
  for (int s = 128; s > 0; s >>= 1) {
    if (t < s) {
      if (sv[t + s] > sv[t] || (sv[t + s] == sv[t] && si[t + s] < si[t])) { sv[t] = sv[t + s]; si[t] = si[t + s]; }
    }
    __syncthreads();
  }
  if (t == 0) nn2[(long)b * NPTS + m] = si[0];
}

__global__ void k_fill_dustbin(float* P, const float* dptr) {
  long i = blockIdx.x * 256L + threadIdx.x;
  long tot = (long)BATCH * NBD;
  if (i >= tot) return;
  int b = (int)(i / NBD), j = (int)(i % NBD);
  float d = *dptr;
  float* base = P + (long)b * NBD * NBD;
  base[(long)j * NBD + (NBD - 1)] = d;
  base[(long)(NBD - 1) * NBD + j] = d;
}

__global__ void k_sink_u(const float* P, const float* v, float* u) {
  __shared__ float sh[256];
  int b = blockIdx.y, i = blockIdx.x, t = threadIdx.x;
  const float* p = P + ((long)b * NBD + i) * NBD;
  const float* vv = v + (long)b * NBD;
  float s = 0.f;
  for (int j = t; j < NBD; j += 256) s += p[j] * vv[j];
  s = blk_sum(s, sh);
  if (t == 0) {
    float r = (i < NPTS) ? 1.f : (float)(NPTS + 1);
    u[(long)b * NBD + i] = r / (s + EPS_);
  }
}

__global__ void k_sink_v(const float* P, const float* u, float* v) {
  __shared__ float sh[256];
  int b = blockIdx.y, j = blockIdx.x, t = threadIdx.x;
  const float* p = P + (long)b * NBD * NBD + j;
  const float* uu = u + (long)b * NBD;
  float s = 0.f;
  for (int i = t; i < NBD; i += 256) s += p[(long)i * NBD] * uu[i];
  s = blk_sum(s, sh);
  if (t == 0) {
    float c = (j < NPTS) ? 1.f : (float)(NPTS + 1);
    v[(long)b * NBD + j] = c / (s + EPS_);
  }
}

__global__ void k_sink_finish(const float* P, const float* u, const float* v, float* out) {
  long i = blockIdx.x * 256L + threadIdx.x;
  long per = (long)NBD * NBD;
  long tot = (long)BATCH * per;
  if (i >= tot) return;
  int b = (int)(i / per);
  long r = i % per;
  int ii = (int)(r / NBD), jj = (int)(r % NBD);
  out[i] = P[i] * u[(long)b * NBD + ii] * v[(long)b * NBD + jj];
}

// ---------------------------------------------------------------------------
// Host-side param structures (JAX pytree flatten order: dict keys sorted)
// ---------------------------------------------------------------------------
struct ConvP { const float* b; const float* w; };
struct BnP   { const float* b; const float* g; const float* rm; const float* rv; };
struct AttnP { BnP cat_bn; ConvP cat_c1, cat_c2, k, mh, q, v; };
struct ConfP { BnP bn1, bn2; ConvP c1, c2, shot; };
struct BlockP { BnP clu_bn; ConvP clu_c1, clu_c2; ConfP conf; AttnP cross, down, selfp, up; };

struct Cur {
  void* const* in; int i;
  const float* nx() { return (const float*)in[i++]; }
};
static ConvP getConv(Cur& c) { ConvP p; p.b = c.nx(); p.w = c.nx(); return p; }
static BnP getBn(Cur& c) { BnP p; p.b = c.nx(); p.g = c.nx(); p.rm = c.nx(); p.rv = c.nx(); return p; }
static AttnP getAttn(Cur& c) {
  AttnP a; a.cat_bn = getBn(c); a.cat_c1 = getConv(c); a.cat_c2 = getConv(c);
  a.k = getConv(c); a.mh = getConv(c); a.q = getConv(c); a.v = getConv(c); return a;
}
static ConfP getConf(Cur& c) {
  ConfP f; f.bn1 = getBn(c); f.bn2 = getBn(c); f.c1 = getConv(c); f.c2 = getConv(c); f.shot = getConv(c); return f;
}
static BlockP getBlock(Cur& c) {
  BlockP b; b.clu_bn = getBn(c); b.clu_c1 = getConv(c); b.clu_c2 = getConv(c);
  b.conf = getConf(c); b.cross = getAttn(c); b.down = getAttn(c); b.selfp = getAttn(c); b.up = getAttn(c);
  return b;
}

// ---------------------------------------------------------------------------
// Workspace
// ---------------------------------------------------------------------------
struct WS {
  float *X1T, *X2T, *XS1, *XS2, *X2G;
  float *D0N, *D1N, *D0T, *D1T, *A0, *A1;
  float *DIS, *PBUF;
  float *TQ, *TK, *TV, *TADD, *TCAT, *TH1, *TS;
  float *SC1, *SC2, *CONC, *CLU, *CLU2, *CT1, *CT2, *CT3;
  float *WTMP, *WBUF, *UV, *VV;
  float *V0, *V1, *RATIO, *SF, *SEL, *MSC, *MEAN;
  int *NN1, *NN2, *S1, *S2;
};

// ---------------------------------------------------------------------------
// GEMM launch wrapper
// ---------------------------------------------------------------------------
struct GemmCfg {
  const float* A = nullptr; const float* Bm = nullptr; float* Cc = nullptr;
  int M = 0, N = 0, K = 0, lda = 0, ldb = 0, ldc = 0;
  long sAb = 0, sBb = 0, sCb = 0; int batch = 1;
  int ZH = 1; long sAh = 0, sBh = 0, sCh = 0;
  const float* bias = nullptr;
  const float* res = nullptr; int ldr = 0; long sRb = 0;
  float alpha = 1.f; int transA = 0, transB = 0, relu = 0, accum = 0, dist = 0;
};
static inline void gemm(hipStream_t st, const GemmCfg& g) {
  dim3 gr((g.N + BN - 1) / BN, (g.M + BM - 1) / BM, g.batch * g.ZH);
  k_gemm<<<gr, 128, 0, st>>>(g.A, g.Bm, g.Cc, g.bias, g.res,
      g.M, g.N, g.K, g.lda, g.ldb, g.ldc, g.ldr,
      g.sAb, g.sAh, g.sBb, g.sBh, g.sCb, g.sCh, g.sRb,
      g.ZH, g.transA, g.transB, g.alpha, g.relu, g.accum, g.dist);
}

static void conv1x1(hipStream_t st, const ConvP& p, const float* X, long sX, float* Y, long sY,
                    int co, int ci, int n, const float* res = nullptr, long sRes = 0, int accum = 0) {
  GemmCfg g; g.A = p.w; g.Bm = X; g.Cc = Y; g.M = co; g.N = n; g.K = ci;
  g.lda = ci; g.ldb = n; g.ldc = n; g.sBb = sX; g.sCb = sY; g.batch = BATCH; g.bias = p.b;
  if (res) { g.res = res; g.ldr = n; g.sRb = sRes; }
  g.accum = accum;
  gemm(st, g);
}

static void copy_ch(hipStream_t st, float* dst, const float* src, int ch, int n, long sD, long sS) {
  long chn = (long)ch * n, tot = (long)BATCH * chn;
  k_copy_ch<<<(unsigned)((tot + 255) / 256), 256, 0, st>>>(dst, src, chn, sD, sS, tot);
}

static void bnrelu(hipStream_t st, float* X, const BnP& p, int ch, int n) {
  long tot = (long)BATCH * ch * n;
  k_bnrelu<<<(unsigned)((tot + 255) / 256), 256, 0, st>>>(X, p.g, p.b, p.rm, p.rv, ch, n, tot);
}

// ---------------------------------------------------------------------------
// Attention (updates D1 in place: D1 += MLP(cat(D1, MHAttn(D1, D2))))
// ---------------------------------------------------------------------------
static void attention(hipStream_t st, WS& w, const AttnP& P,
                      float* D1, int n1, long sD1,
                      const float* D2, int n2, long sD2,
                      const float* wv) {
  GemmCfg g;
  // q = Wq*D1 + bq
  g = GemmCfg(); g.A = P.q.w; g.Bm = D1; g.Cc = w.TQ; g.M = CH; g.N = n1; g.K = CH;
  g.lda = CH; g.ldb = n1; g.ldc = n1; g.sBb = sD1; g.sCb = (long)CH * n1; g.batch = BATCH; g.bias = P.q.b;
  gemm(st, g);
  // k = Wk*D2 + bk
  g = GemmCfg(); g.A = P.k.w; g.Bm = D2; g.Cc = w.TK; g.M = CH; g.N = n2; g.K = CH;
  g.lda = CH; g.ldb = n2; g.ldc = n2; g.sBb = sD2; g.sCb = (long)CH * n2; g.batch = BATCH; g.bias = P.k.b;
  gemm(st, g);
  // v = Wv*D2 + bv
  g.A = P.v.w; g.bias = P.v.b; g.Cc = w.TV;
  gemm(st, g);
  if (wv) {
    long tot = (long)BATCH * CH * n2;
    k_mulcols<<<(unsigned)((tot + 255) / 256), 256, 0, st>>>(w.TV, wv, n2, tot);
  }
  // scores[b,h] = (Q^T K)/8
  g = GemmCfg(); g.A = w.TQ; g.Bm = w.TK; g.Cc = w.TS; g.M = n1; g.N = n2; g.K = HDIM;
  g.lda = n1; g.ldb = n2; g.ldc = n2; g.transA = 1; g.alpha = 0.125f;
  g.batch = BATCH; g.ZH = NH;
  g.sAb = (long)CH * n1; g.sAh = (long)HDIM * n1;
  g.sBb = (long)CH * n2; g.sBh = (long)HDIM * n2;
  g.sCb = (long)NH * n1 * n2; g.sCh = (long)n1 * n2;
  gemm(st, g);
  k_softmax_rows<<<(unsigned)(BATCH * NH * n1), 256, 0, st>>>(w.TS, n2, n2);
  // add[b,h] = V * S^T
  g = GemmCfg(); g.A = w.TV; g.Bm = w.TS; g.Cc = w.TADD; g.M = HDIM; g.N = n1; g.K = n2;
  g.lda = n2; g.ldb = n2; g.ldc = n1; g.transB = 1;
  g.batch = BATCH; g.ZH = NH;
  g.sAb = (long)CH * n2; g.sAh = (long)HDIM * n2;
  g.sBb = (long)NH * n1 * n2; g.sBh = (long)n1 * n2;
  g.sCb = (long)CH * n1; g.sCh = (long)HDIM * n1;
  gemm(st, g);
  // mh conv -> lower half of TCAT; upper half = D1
  g = GemmCfg(); g.A = P.mh.w; g.Bm = w.TADD; g.Cc = w.TCAT + (long)CH * n1; g.M = CH; g.N = n1; g.K = CH;
  g.lda = CH; g.ldb = n1; g.ldc = n1; g.sBb = (long)CH * n1; g.sCb = (long)2 * CH * n1;
  g.batch = BATCH; g.bias = P.mh.b;
  gemm(st, g);
  copy_ch(st, w.TCAT, D1, CH, n1, (long)2 * CH * n1, sD1);
  // cat_c1
  g = GemmCfg(); g.A = P.cat_c1.w; g.Bm = w.TCAT; g.Cc = w.TH1; g.M = 2 * CH; g.N = n1; g.K = 2 * CH;
  g.lda = 2 * CH; g.ldb = n1; g.ldc = n1; g.sBb = (long)2 * CH * n1; g.sCb = (long)2 * CH * n1;
  g.batch = BATCH; g.bias = P.cat_c1.b;
  gemm(st, g);
  bnrelu(st, w.TH1, P.cat_bn, 2 * CH, n1);
  // cat_c2 + residual D1 -> D1
  g = GemmCfg(); g.A = P.cat_c2.w; g.Bm = w.TH1; g.Cc = D1; g.M = CH; g.N = n1; g.K = 2 * CH;
  g.lda = 2 * CH; g.ldb = n1; g.ldc = n1; g.sBb = (long)2 * CH * n1; g.sCb = sD1;
  g.batch = BATCH; g.bias = P.cat_c2.b;
  g.res = D1; g.ldr = n1; g.sRb = sD1;
  gemm(st, g);
}

// ---------------------------------------------------------------------------
// Seeding (mutual-NN + NMS + top-128)
// ---------------------------------------------------------------------------
static void seeding(hipStream_t st, WS& w, const int* nn1, const int* nn2,
                    const float* score, float confbar) {
  long tot = (long)BATCH * NPTS;
  unsigned blk = (unsigned)((tot + 255) / 256);
  k_gatherxy<<<blk, 256, 0, st>>>(w.XS2, nn1, w.X2G);
  k_set<<<1, 64, 0, st>>>(w.MEAN, 0.f, 2 * BATCH);
  dim3 gp(NPTS, BATCH);
  k_pdist_sum<<<gp, 256, 0, st>>>(w.XS1, w.MEAN, 0);
  k_pdist_sum<<<gp, 256, 0, st>>>(w.X2G, w.MEAN, BATCH);
  k_mutual<<<blk, 256, 0, st>>>(nn1, nn2, score, w.SF);
  k_ismax<<<gp, 256, 0, st>>>(w.SF, w.XS1, w.X2G, w.MEAN, w.SEL, confbar);
  k_topk<<<BATCH, 256, 0, st>>>(w.SEL, nn1, w.S1, w.S2);
}

// ---------------------------------------------------------------------------
// Hybrid block
// ---------------------------------------------------------------------------
static void hybrid(hipStream_t st, WS& w, const BlockP& bp) {
  const long cs = (long)CH * SEED, cn = (long)CH * NPTS;
  long tot = (long)BATCH * CH * SEED;
  unsigned blk = (unsigned)((tot + 255) / 256);
  k_gather_cols<<<blk, 256, 0, st>>>(w.SC1, w.A0, w.S1, NPTS, cn);
  k_gather_cols<<<blk, 256, 0, st>>>(w.SC2, w.A1, w.S2, NPTS, cn);
  attention(st, w, bp.down, w.SC1, SEED, cs, w.A0, NPTS, cn, nullptr);
  attention(st, w, bp.down, w.SC2, SEED, cs, w.A1, NPTS, cn, nullptr);
  // cluster mixing
  copy_ch(st, w.CONC, w.SC1, CH, SEED, 2 * cs, cs);
  copy_ch(st, w.CONC + cs, w.SC2, CH, SEED, 2 * cs, cs);
  conv1x1(st, bp.clu_c1, w.CONC, 2 * cs, w.CLU, 2 * cs, 2 * CH, 2 * CH, SEED);
  bnrelu(st, w.CLU, bp.clu_bn, 2 * CH, SEED);
  conv1x1(st, bp.clu_c2, w.CLU, 2 * cs, w.CLU2, 2 * cs, 2 * CH, 2 * CH, SEED);
  copy_ch(st, w.SC1, w.CLU2, CH, SEED, cs, 2 * cs);
  copy_ch(st, w.SC2, w.CLU2 + cs, CH, SEED, cs, 2 * cs);
  attention(st, w, bp.cross, w.SC1, SEED, cs, w.CLU2 + cs, SEED, 2 * cs, nullptr);
  attention(st, w, bp.cross, w.SC2, SEED, cs, w.CLU2, SEED, 2 * cs, nullptr);
  attention(st, w, bp.selfp, w.SC1, SEED, cs, w.SC1, SEED, cs, nullptr);
  attention(st, w, bp.selfp, w.SC2, SEED, cs, w.SC2, SEED, cs, nullptr);
  // confidence head (pointcn)
  copy_ch(st, w.CONC, w.SC1, CH, SEED, 2 * cs, cs);
  copy_ch(st, w.CONC + cs, w.SC2, CH, SEED, 2 * cs, cs);
  k_inorm_bnrelu<<<BATCH * 2 * CH, 256, 0, st>>>(w.CONC, w.CT1,
      bp.conf.bn1.g, bp.conf.bn1.b, bp.conf.bn1.rm, bp.conf.bn1.rv, 2 * CH, SEED);
  conv1x1(st, bp.conf.c1, w.CT1, 2 * cs, w.CT2, 2 * cs, 2 * CH, 2 * CH, SEED);
  k_inorm_bnrelu<<<BATCH * 2 * CH, 256, 0, st>>>(w.CT2, w.CT3,
      bp.conf.bn2.g, bp.conf.bn2.b, bp.conf.bn2.rm, bp.conf.bn2.rv, 2 * CH, SEED);
  conv1x1(st, bp.conf.c2, w.CT3, 2 * cs, w.WTMP, SEED, 1, 2 * CH, SEED);
  conv1x1(st, bp.conf.shot, w.CONC, 2 * cs, w.WTMP, SEED, 1, 2 * CH, SEED, nullptr, 0, /*accum*/1);
  long wt = (long)BATCH * SEED;
  k_sigmoid<<<(unsigned)((wt + 255) / 256), 256, 0, st>>>(w.WTMP, w.WBUF, wt);
  // up attentions with weighted values
  attention(st, w, bp.up, w.A0, NPTS, cn, w.SC1, SEED, cs, w.WBUF);
  attention(st, w, bp.up, w.A1, NPTS, cn, w.SC2, SEED, cs, w.WBUF);
}

// ---------------------------------------------------------------------------
// Sinkhorn: cost matrix already in PBUF[0..N)^2; add dustbins, softmax rows,
// iterate u/v, expand into `out`.
// ---------------------------------------------------------------------------
static void sinkhorn(hipStream_t st, WS& w, const float* dust, int iters, float* out) {
  long tdb = (long)BATCH * NBD;
  k_fill_dustbin<<<(unsigned)((tdb + 255) / 256), 256, 0, st>>>(w.PBUF, dust);
  k_softmax_rows<<<(unsigned)(BATCH * NBD), 256, 0, st>>>(w.PBUF, NBD, NBD);
  k_set<<<(unsigned)((tdb + 255) / 256), 256, 0, st>>>(w.UV, 1.f, tdb);
  k_set<<<(unsigned)((tdb + 255) / 256), 256, 0, st>>>(w.VV, 1.f, tdb);
  dim3 gp(NBD, BATCH);
  for (int it = 0; it < iters; ++it) {
    k_sink_u<<<gp, 256, 0, st>>>(w.PBUF, w.VV, w.UV);
    k_sink_v<<<gp, 256, 0, st>>>(w.PBUF, w.UV, w.VV);
  }
  long tot = (long)BATCH * NBD * NBD;
  k_sink_finish<<<(unsigned)((tot + 255) / 256), 256, 0, st>>>(w.PBUF, w.UV, w.VV, out);
}

// ---------------------------------------------------------------------------
// Entry point
// ---------------------------------------------------------------------------
extern "C" void kernel_launch(void* const* d_in, const int* in_sizes, int n_in,
                              void* d_out, int out_size, void* d_ws, size_t ws_size,
                              hipStream_t stream) {
  (void)in_sizes; (void)n_in; (void)out_size;
  hipStream_t st = stream;
  const float* kpts0 = (const float*)d_in[0];
  const float* kpts1 = (const float*)d_in[1];
  const float* desc0 = (const float*)d_in[2];
  const float* desc1 = (const float*)d_in[3];

  // --- params (pytree flatten: sorted dict keys, lists in order) ---
  Cur cur{d_in, 4};
  BlockP blocks[NLAYERS];
  for (int i = 0; i < NLAYERS; ++i) blocks[i] = getBlock(cur);
  const float* dustbin = cur.nx();
  ConvP finalP = getConv(cur);
  const float* mid_dustbin = cur.nx();
  ConvP midP = getConv(cur);
  ConvP pc[5]; BnP pb[4];
  pc[0] = getConv(cur); pb[0] = getBn(cur);
  pc[1] = getConv(cur); pb[1] = getBn(cur);
  pc[2] = getConv(cur); pb[2] = getBn(cur);
  pc[3] = getConv(cur); pb[3] = getBn(cur);
  pc[4] = getConv(cur);

  // --- workspace carve ---
  float* base = (float*)d_ws;
  size_t off = 0;
  auto al = [&](size_t nf) { float* p = base + off; off += nf; return p; };
  WS w;
  const long CN = (long)CH * NPTS, CS = (long)CH * SEED;
  w.X1T = al((long)BATCH * 2 * NPTS);   w.X2T = al((long)BATCH * 2 * NPTS);
  w.XS1 = al((long)BATCH * NPTS * 2);   w.XS2 = al((long)BATCH * NPTS * 2);
  w.X2G = al((long)BATCH * NPTS * 2);
  w.D0N = al((long)BATCH * NPTS * CH);  w.D1N = al((long)BATCH * NPTS * CH);
  w.D0T = al((long)BATCH * CN);         w.D1T = al((long)BATCH * CN);
  w.A0 = al((long)BATCH * CN);          w.A1 = al((long)BATCH * CN);
  w.DIS = al((long)BATCH * NPTS * NPTS);
  w.PBUF = al((long)BATCH * NBD * NBD);
  w.TQ = al((long)BATCH * CN); w.TK = al((long)BATCH * CN);
  w.TV = al((long)BATCH * CN); w.TADD = al((long)BATCH * CN);
  w.TCAT = al((long)BATCH * 2 * CN);    w.TH1 = al((long)BATCH * 2 * CN);
  w.TS = al((long)BATCH * NH * NPTS * SEED);
  w.SC1 = al((long)BATCH * CS);         w.SC2 = al((long)BATCH * CS);
  w.CONC = al((long)BATCH * 2 * CS);    w.CLU = al((long)BATCH * 2 * CS);
  w.CLU2 = al((long)BATCH * 2 * CS);
  w.CT1 = al((long)BATCH * 2 * CS); w.CT2 = al((long)BATCH * 2 * CS); w.CT3 = al((long)BATCH * 2 * CS);
  w.WTMP = al((long)BATCH * SEED);      w.WBUF = al((long)BATCH * SEED);
  w.UV = al((long)BATCH * NBD);         w.VV = al((long)BATCH * NBD);
  w.V0 = al((long)BATCH * NPTS); w.V1 = al((long)BATCH * NPTS);
  w.RATIO = al((long)BATCH * NPTS); w.SF = al((long)BATCH * NPTS);
  w.SEL = al((long)BATCH * NPTS); w.MSC = al((long)BATCH * NPTS);
  w.MEAN = al(2 * BATCH);
  w.NN1 = (int*)al((long)BATCH * NPTS); w.NN2 = (int*)al((long)BATCH * NPTS);
  w.S1 = (int*)al((long)BATCH * SEED);  w.S2 = (int*)al((long)BATCH * SEED);
  if (off * sizeof(float) > ws_size) return;  // insufficient scratch

  // --- coords + normalized descriptors ---
  long tbn = (long)BATCH * NPTS;
  unsigned bbn = (unsigned)((tbn + 255) / 256);
  k_xt<<<bbn, 256, 0, st>>>(kpts0, w.X1T, w.XS1);
  k_xt<<<bbn, 256, 0, st>>>(kpts1, w.X2T, w.XS2);
  dim3 gnd(NPTS, BATCH);
  k_normdesc<<<gnd, 256, 0, st>>>(desc0, w.D0N, w.D0T);
  k_normdesc<<<gnd, 256, 0, st>>>(desc1, w.D1N, w.D1T);

  // --- distance matrix (WMMA GEMM, dist epilogue) + top-2/argmin + ratio ---
  {
    GemmCfg g; g.A = w.D0N; g.Bm = w.D1N; g.Cc = w.DIS;
    g.M = NPTS; g.N = NPTS; g.K = CH;
    g.lda = CH; g.ldb = CH; g.ldc = NPTS; g.transB = 1; g.dist = 1;
    g.sAb = (long)NPTS * CH; g.sBb = (long)NPTS * CH; g.sCb = (long)NPTS * NPTS; g.batch = BATCH;
    gemm(st, g);
  }
  k_top2<<<gnd, 256, 0, st>>>(w.DIS, w.V0, w.V1, w.NN1);
  k_colargmin<<<gnd, 256, 0, st>>>(w.DIS, w.NN2);
  k_ratio<<<bbn, 256, 0, st>>>(w.V0, w.V1, w.RATIO, tbn);
  seeding(st, w, w.NN1, w.NN2, w.RATIO, 1.0f);

  // --- positional encoding chains: A = desc^T + posenc(x) ---
  auto pos_chain = [&](const float* XT, const float* resid, float* out) {
    conv1x1(st, pc[0], XT, 2L * NPTS, w.TQ, 32L * NPTS, 32, 2, NPTS);
    bnrelu(st, w.TQ, pb[0], 32, NPTS);
    conv1x1(st, pc[1], w.TQ, 32L * NPTS, w.TK, 64L * NPTS, 64, 32, NPTS);
    bnrelu(st, w.TK, pb[1], 64, NPTS);
    conv1x1(st, pc[2], w.TK, 64L * NPTS, w.TQ, 128L * NPTS, 128, 64, NPTS);
    bnrelu(st, w.TQ, pb[2], 128, NPTS);
    conv1x1(st, pc[3], w.TQ, 128L * NPTS, w.TK, 256L * NPTS, 256, 128, NPTS);
    bnrelu(st, w.TK, pb[3], 256, NPTS);
    conv1x1(st, pc[4], w.TK, 256L * NPTS, out, CN, CH, 256, NPTS, resid, CN);
  };
  pos_chain(w.X1T, w.D0T, w.A0);
  pos_chain(w.X2T, w.D1T, w.A1);

  // --- layers ---
  for (int i = 0; i < NLAYERS; ++i) {
    if (i == MIDL) {
      // mid_final conv (out-of-place then copy back)
      conv1x1(st, midP, w.A0, CN, w.TQ, CN, CH, CH, NPTS);
      copy_ch(st, w.A0, w.TQ, CH, NPTS, CN, CN);
      conv1x1(st, midP, w.A1, CN, w.TK, CN, CH, CH, NPTS);
      copy_ch(st, w.A1, w.TK, CH, NPTS, CN, CN);
      // p_mid = sink(a0^T a1, mid_dustbin, 10)
      GemmCfg g; g.A = w.A0; g.Bm = w.A1; g.Cc = w.PBUF;
      g.M = NPTS; g.N = NPTS; g.K = CH;
      g.lda = NPTS; g.ldb = NPTS; g.ldc = NBD; g.transA = 1;
      g.sAb = CN; g.sBb = CN; g.sCb = (long)NBD * NBD; g.batch = BATCH;
      gemm(st, g);
      sinkhorn(st, w, mid_dustbin, 10, w.PBUF);
      k_rowmax_arg<<<gnd, 256, 0, st>>>(w.PBUF, w.MSC, w.NN1);
      k_colmax_arg<<<gnd, 256, 0, st>>>(w.PBUF, w.NN2);
      seeding(st, w, w.NN1, w.NN2, w.MSC, 0.1f);
    }
    hybrid(st, w, blocks[i]);
  }

  // --- final projection + correlation + sinkhorn(100) -> d_out ---
  conv1x1(st, finalP, w.A0, CN, w.TQ, CN, CH, CH, NPTS);
  conv1x1(st, finalP, w.A1, CN, w.TK, CN, CH, CH, NPTS);
  {
    GemmCfg g; g.A = w.TQ; g.Bm = w.TK; g.Cc = w.PBUF;
    g.M = NPTS; g.N = NPTS; g.K = CH;
    g.lda = NPTS; g.ldb = NPTS; g.ldc = NBD; g.transA = 1;
    g.alpha = 1.0f / 16.0f;  // / sqrt(C)
    g.sAb = CN; g.sBb = CN; g.sCb = (long)NBD * NBD; g.batch = BATCH;
    gemm(st, g);
  }
  sinkhorn(st, w, dustbin, 100, (float*)d_out);
}
